// NetVLAD_58746562675115
// MI455X (gfx1250) — matrix operand order repaired and checked by reading, not compiled, not run on hardware
//
#include <hip/hip_runtime.h>
#include <math.h>

// ---- types ------------------------------------------------------------
typedef __bf16 bf_t;
typedef __attribute__((ext_vector_type(16))) __bf16 v16bf;
typedef __attribute__((ext_vector_type(8)))  __bf16 v8bf;
typedef __attribute__((ext_vector_type(4)))  __bf16 v4bf;
typedef __attribute__((ext_vector_type(2)))  __bf16 v2bf;
typedef __attribute__((ext_vector_type(8)))  float  v8f;

// ---- problem sizes (fixed by reference) -------------------------------
#define NB 8
#define TB 2048
#define DB 512
#define KB 64
#define NT (NB * TB)          // 16384 descriptor rows
#define EPSF 1e-12f

static __device__ __forceinline__ float wave_sum32(float v) {
#pragma unroll
  for (int m = 16; m >= 1; m >>= 1) v += __shfl_xor(v, m, 32);
  return v;
}

// A 16x32 bf16 fragment: lane holds row (lane&15); VGPR u holds the K pair
// K = 16*(lane>>4) + 2*u. Pairs are K-consecutive -> dword loads.
static __device__ __forceinline__ v16bf load_frag_a(const bf_t* __restrict__ p, int h) {
  v16bf f;
#pragma unroll
  for (int u = 0; u < 8; ++u) {
    const int kk = (h << 4) + (u << 1);
    v2bf t = *reinterpret_cast<const v2bf*>(p + kk);
    f[2 * u] = t[0]; f[2 * u + 1] = t[1];
  }
  return f;
}

// B 32x16 bf16 fragment: lane holds col (lane&15), half h holds K=16h..16h+15,
// 2 K-consecutive values per VGPR. p must already include the +16h offset.
static __device__ __forceinline__ v16bf load_frag_b(const bf_t* p) {
  v16bf f;
#pragma unroll
  for (int u = 0; u < 8; ++u) {
    v2bf t = *reinterpret_cast<const v2bf*>(p + 2 * u);
    f[2 * u] = t[0]; f[2 * u + 1] = t[1];
  }
  return f;
}

// =======================================================================
// K1: L2-normalize x along D, emit bf16 row-major xn[nt][d] and
//     transposed xnT[n][d][t]. One wave per (n,t) row.
// =======================================================================
__global__ __launch_bounds__(256) void nv_normalize(const float* __restrict__ x,
                                                    bf_t* __restrict__ xn,
                                                    bf_t* __restrict__ xnT) {
  const int wid  = (blockIdx.x * blockDim.x + threadIdx.x) >> 5;  // row 0..NT-1
  const int lane = threadIdx.x & 31;
  const float* row = x + (size_t)wid * DB;

  float4 v[4];
  float s = 0.f;
#pragma unroll
  for (int i = 0; i < 4; ++i) {
    v[i] = *reinterpret_cast<const float4*>(row + 4 * lane + 128 * i);
    s += v[i].x * v[i].x + v[i].y * v[i].y + v[i].z * v[i].z + v[i].w * v[i].w;
  }
  s = wave_sum32(s);
  const float inv = 1.0f / fmaxf(sqrtf(s), EPSF);

  const int n = wid >> 11;            // wid / TB
  const int t = wid & (TB - 1);
  bf_t* orow = xn + (size_t)wid * DB;
#pragma unroll
  for (int i = 0; i < 4; ++i) {
    const int d = 4 * lane + 128 * i;
    v4bf o;
    o[0] = (bf_t)(v[i].x * inv); o[1] = (bf_t)(v[i].y * inv);
    o[2] = (bf_t)(v[i].z * inv); o[3] = (bf_t)(v[i].w * inv);
    *reinterpret_cast<v4bf*>(orow + d) = o;
    // transposed copy for GEMM2 A-matrix (t contiguous per d-row)
    xnT[((size_t)(n * DB + d + 0)) * TB + t] = o[0];
    xnT[((size_t)(n * DB + d + 1)) * TB + t] = o[1];
    xnT[((size_t)(n * DB + d + 2)) * TB + t] = o[2];
    xnT[((size_t)(n * DB + d + 3)) * TB + t] = o[3];
  }
}

// =======================================================================
// K2: W (fp32 [K,D]) -> bf16
// =======================================================================
__global__ __launch_bounds__(256) void nv_wconv(const float* __restrict__ W,
                                                bf_t* __restrict__ wbf) {
  const int i = blockIdx.x * blockDim.x + threadIdx.x;  // 0..K*D-1 exact
  wbf[i] = (bf_t)W[i];
}

// =======================================================================
// K3: logits = xn @ W^T + b  (WMMA bf16), softmax over K=64 in registers,
//     store a as bf16 [n][k][t] and accumulate asum[n][k] = sum_t a.
//     One wave per 16-row tile of xn; 4 accumulators cover all 64 clusters.
//     W (64KB bf16) staged once per block into LDS with CDNA5 async copies.
// =======================================================================
__global__ __launch_bounds__(256) void nv_logits(const bf_t* __restrict__ xn,
                                                 const bf_t* __restrict__ wbf_g,
                                                 const float* __restrict__ bias,
                                                 bf_t* __restrict__ a_ws,
                                                 float* __restrict__ asum) {
  __shared__ bf_t wlds[KB * DB];                 // 64 KB of the 320 KB WGP LDS

  // ---- async stage W -> LDS (ASYNCcnt path): 256 threads x 256 B -------
  {
    const int tid = threadIdx.x;
    // flat shared pointer: low 32 bits are the LDS byte offset (ISA 10.2)
    unsigned ldsoff = (unsigned)(uintptr_t)(&wlds[0]) + (unsigned)tid * 256u;
    const char* gsrc = (const char*)wbf_g + (size_t)tid * 256;
#pragma unroll
    for (int i = 0; i < 16; ++i) {
      asm volatile("global_load_async_to_lds_b128 %0, %1, off offset:%2"
                   :: "v"(ldsoff), "v"(gsrc), "i"(i * 16)
                   : "memory");
    }
    asm volatile("s_wait_asynccnt 0" ::: "memory");
    __syncthreads();
  }

  const int wid  = (blockIdx.x * blockDim.x + threadIdx.x) >> 5;  // 0..1023
  const int lane = threadIdx.x & 31;
  const int l15  = lane & 15;
  const int h    = lane >> 4;
  const int n    = wid >> 7;            // wid / (TB/16)
  const int t0   = (wid & 127) << 4;
  const int row0 = wid << 4;            // global row in [NT)

  v8f c[4] = {};
  const bf_t* ap = xn + (size_t)(row0 + l15) * DB;

  for (int d0 = 0; d0 < DB; d0 += 32) {
    // issue all fragment loads first (distinct regs -> one clause, no
    // per-WMMA waits), then run the 4 WMMAs back-to-back
    v16bf af = load_frag_a(ap + d0, h);
    v16bf b0 = load_frag_b(wlds + (size_t)(l15 +  0) * DB + d0 + (h << 4));
    v16bf b1 = load_frag_b(wlds + (size_t)(l15 + 16) * DB + d0 + (h << 4));
    v16bf b2 = load_frag_b(wlds + (size_t)(l15 + 32) * DB + d0 + (h << 4));
    v16bf b3 = load_frag_b(wlds + (size_t)(l15 + 48) * DB + d0 + (h << 4));
    c[0] = __builtin_amdgcn_wmma_f32_16x16x32_bf16(false, af, false, b0, (short)0, c[0], false, false);
    c[1] = __builtin_amdgcn_wmma_f32_16x16x32_bf16(false, af, false, b1, (short)0, c[1], false, false);
    c[2] = __builtin_amdgcn_wmma_f32_16x16x32_bf16(false, af, false, b2, (short)0, c[2], false, false);
    c[3] = __builtin_amdgcn_wmma_f32_16x16x32_bf16(false, af, false, b3, (short)0, c[3], false, false);
  }

  // bias (per column)
#pragma unroll
  for (int g = 0; g < 4; ++g) {
    const float bg = bias[(g << 4) + l15];
#pragma unroll
    for (int r = 0; r < 8; ++r) c[g][r] += bg;
  }

  // softmax over 64 clusters per row; row m = 8*h + r lives in the 16 lanes
  // of half h (cols l15 across lanes x 4 accumulator groups)
  float av[4][8];
#pragma unroll
  for (int r = 0; r < 8; ++r) {
    float mx = fmaxf(fmaxf(c[0][r], c[1][r]), fmaxf(c[2][r], c[3][r]));
#pragma unroll
    for (int msk = 8; msk >= 1; msk >>= 1) mx = fmaxf(mx, __shfl_xor(mx, msk, 32));
    float sm = 0.f;
#pragma unroll
    for (int g = 0; g < 4; ++g) { av[g][r] = __expf(c[g][r] - mx); sm += av[g][r]; }
#pragma unroll
    for (int msk = 8; msk >= 1; msk >>= 1) sm += __shfl_xor(sm, msk, 32);
    const float isum = 1.0f / sm;
#pragma unroll
    for (int g = 0; g < 4; ++g) av[g][r] *= isum;
  }

  // store a -> [n][k][t] (8 consecutive t per lane = one 16B store) and asum
#pragma unroll
  for (int g = 0; g < 4; ++g) {
    const int col = (g << 4) + l15;
    v8bf st;
    float p = 0.f;
#pragma unroll
    for (int r = 0; r < 8; ++r) { st[r] = (bf_t)av[g][r]; p += av[g][r]; }
    *reinterpret_cast<v8bf*>(a_ws + ((size_t)(n * KB + col)) * TB + t0 + (h << 3)) = st;
    p += __shfl_xor(p, 16, 32);
    if (h == 0) atomicAdd(asum + n * KB + col, p);
  }
}

// =======================================================================
// K4: vlad'[n][d][k] = sum_t xnT[n][d][t] * a[n][k][t]  (WMMA bf16),
//     then subtract asum[n][k]*centroids[k][d], store fp32 [n][k][d].
//     One wave per (n, 16-d-row tile); 4 accumulators cover K=64.
// =======================================================================
__global__ __launch_bounds__(256) void nv_vlad(const bf_t* __restrict__ xnT,
                                               const bf_t* __restrict__ a_ws,
                                               const float* __restrict__ asum,
                                               const float* __restrict__ cent,
                                               float* __restrict__ vlad) {
  const int wid  = (blockIdx.x * blockDim.x + threadIdx.x) >> 5;  // 0..255
  const int lane = threadIdx.x & 31;
  const int l15  = lane & 15;
  const int h    = lane >> 4;
  const int n    = wid >> 5;
  const int d0   = (wid & 31) << 4;

  v8f c[4] = {};
  const bf_t* ap = xnT + ((size_t)(n * DB + d0 + l15)) * TB;
  const bf_t* bbase = a_ws + ((size_t)(n * KB + l15)) * TB + (h << 4);

  for (int t0 = 0; t0 < TB; t0 += 32) {
    v16bf af = load_frag_a(ap + t0, h);
    v16bf b0 = load_frag_b(bbase + (size_t)( 0 * TB) + t0);
    v16bf b1 = load_frag_b(bbase + (size_t)(16 * TB) + t0);
    v16bf b2 = load_frag_b(bbase + (size_t)(32 * TB) + t0);
    v16bf b3 = load_frag_b(bbase + (size_t)(48 * TB) + t0);
    c[0] = __builtin_amdgcn_wmma_f32_16x16x32_bf16(false, af, false, b0, (short)0, c[0], false, false);
    c[1] = __builtin_amdgcn_wmma_f32_16x16x32_bf16(false, af, false, b1, (short)0, c[1], false, false);
    c[2] = __builtin_amdgcn_wmma_f32_16x16x32_bf16(false, af, false, b2, (short)0, c[2], false, false);
    c[3] = __builtin_amdgcn_wmma_f32_16x16x32_bf16(false, af, false, b3, (short)0, c[3], false, false);
  }

  // C layout: lane holds col k=(g<<4)+l15, VGPR r holds d-row d0+8h+r
#pragma unroll
  for (int g = 0; g < 4; ++g) {
    const int k = (g << 4) + l15;
    const float sA = asum[n * KB + k];
    const float* crow = cent + (size_t)k * DB + d0 + (h << 3);
    float* orow = vlad + ((size_t)(n * KB + k)) * DB + d0 + (h << 3);
#pragma unroll
    for (int r = 0; r < 8; ++r) orow[r] = c[g][r] - sA * crow[r];
  }
}

// =======================================================================
// K5: intra-normalize each (n,k) row of vlad (512 floats, one wave each),
//     write to out, accumulate per-n global sum of squares.
// =======================================================================
__global__ __launch_bounds__(256) void nv_intra(const float* __restrict__ vlad,
                                                float* __restrict__ out,
                                                float* __restrict__ gs) {
  const int wid  = (blockIdx.x * blockDim.x + threadIdx.x) >> 5;  // 0..511
  const int lane = threadIdx.x & 31;
  const float* row = vlad + (size_t)wid * DB;

  float4 v[4];
  float s = 0.f;
#pragma unroll
  for (int i = 0; i < 4; ++i) {
    v[i] = *reinterpret_cast<const float4*>(row + 4 * lane + 128 * i);
    s += v[i].x * v[i].x + v[i].y * v[i].y + v[i].z * v[i].z + v[i].w * v[i].w;
  }
  s = wave_sum32(s);
  const float inv = 1.0f / fmaxf(sqrtf(s), EPSF);

  float* orow = out + (size_t)wid * DB;
#pragma unroll
  for (int i = 0; i < 4; ++i) {
    float4 o;
    o.x = v[i].x * inv; o.y = v[i].y * inv; o.z = v[i].z * inv; o.w = v[i].w * inv;
    *reinterpret_cast<float4*>(orow + 4 * lane + 128 * i) = o;
  }
  if (lane == 0) atomicAdd(gs + (wid >> 6), s * inv * inv);
}

// =======================================================================
// K6: global L2 normalize each n-row of out
// =======================================================================
__global__ __launch_bounds__(256) void nv_gscale(float* __restrict__ out,
                                                 const float* __restrict__ gs) {
  const int idx = blockIdx.x * blockDim.x + threadIdx.x;  // 0..N*K*D-1 exact
  const float g = gs[idx >> 15];                          // idx / (K*D)
  out[idx] *= 1.0f / fmaxf(sqrtf(g), EPSF);
}

// =======================================================================
// launch
// =======================================================================
extern "C" void kernel_launch(void* const* d_in, const int* in_sizes, int n_in,
                              void* d_out, int out_size, void* d_ws, size_t ws_size,
                              hipStream_t stream) {
  (void)in_sizes; (void)n_in; (void)out_size; (void)ws_size;
  const float* x    = (const float*)d_in[0];  // [N,T,D]
  const float* W    = (const float*)d_in[1];  // [K,D]
  const float* b    = (const float*)d_in[2];  // [K]
  const float* cent = (const float*)d_in[3];  // [K,D]
  float* out = (float*)d_out;                 // [N, K*D]

  char* w = (char*)d_ws;
  size_t off = 0;
  bf_t* xn   = (bf_t*)(w + off); off += (size_t)NT * DB * 2;        // 16.8 MB
  bf_t* xnT  = (bf_t*)(w + off); off += (size_t)NT * DB * 2;        // 16.8 MB
  bf_t* wbf  = (bf_t*)(w + off); off += (size_t)KB * DB * 2;        // 64 KB
  bf_t* aws  = (bf_t*)(w + off); off += (size_t)NB * KB * TB * 2;   // 2 MB
  float* vlad = (float*)(w + off); off += (size_t)NB * KB * DB * 4; // 1 MB
  float* asum = (float*)(w + off); off += (size_t)NB * KB * 4;      // 2 KB
  float* gs   = (float*)(w + off); off += (size_t)NB * 4;           // 32 B

  // zero the atomic accumulators (asum and gs are adjacent)
  hipMemsetAsync(asum, 0, (size_t)(NB * KB + NB) * sizeof(float), stream);

  nv_normalize<<<NT / 8, 256, 0, stream>>>(x, xn, xnT);                 // 2048 blocks
  nv_wconv<<<(KB * DB) / 256, 256, 0, stream>>>(W, wbf);                // 128 blocks
  nv_logits<<<(NT / 16) / 8, 256, 0, stream>>>(xn, wbf, b, aws, asum);  // 128 blocks
  nv_vlad<<<(NB * 32) / 8, 256, 0, stream>>>(xnT, aws, asum, cent, vlad); // 32 blocks
  nv_intra<<<(NB * KB) / 8, 256, 0, stream>>>(vlad, out, gs);           // 64 blocks
  nv_gscale<<<(NB * KB * DB) / 256, 256, 0, stream>>>(out, gs);         // 1024 blocks
}
